// FeatureDifferenceEdgeModel_23510650978557
// MI455X (gfx1250) — matrix operand order repaired and checked by reading, not compiled, not run on hardware
//
#include <hip/hip_runtime.h>
#include <hip/hip_bf16.h>

#define HD   256      // hidden size
#define NP   512      // nodes per batch
#define BB   2        // batch
#define MT   128      // pair-rows (j) per block
#define PAD  8
#define ROWP (HD + PAD)   // padded LDS row, bf16 elements

typedef __bf16 bf16_t;
typedef __attribute__((ext_vector_type(16))) __bf16 v16bf;
typedef __attribute__((ext_vector_type(8)))  __bf16 v8bf;
typedef __attribute__((ext_vector_type(8)))  float  v8f;

// ---------------------------------------------------------------------------
// Small fp32 linear (+optional ReLU): one block per row, one thread per out col.
// ---------------------------------------------------------------------------
__global__ __launch_bounds__(HD)
void lin_kernel(const float* __restrict__ X, const float* __restrict__ W,
                const float* __restrict__ bias, float* __restrict__ Y, int relu)
{
    __shared__ float xs[HD];
    const int row = blockIdx.x;
    const int c   = threadIdx.x;
    xs[c] = X[(size_t)row * HD + c];
    __syncthreads();
    float acc = bias ? bias[c] : 0.f;
#pragma unroll 4
    for (int k = 0; k < HD; ++k)
        acc += xs[k] * W[(size_t)k * HD + c];
    if (relu) acc = acc > 0.f ? acc : 0.f;
    Y[(size_t)row * HD + c] = acc;
}

// ---------------------------------------------------------------------------
// Transpose + convert: Wt[n][k] = (bf16) W[k][n]   (HD x HD)
// ---------------------------------------------------------------------------
__global__ __launch_bounds__(HD)
void tconv_kernel(const float* __restrict__ W, bf16_t* __restrict__ Wt)
{
    const int n = blockIdx.x;
    const int k = threadIdx.x;
    Wt[(size_t)n * HD + k] = (bf16_t)W[(size_t)k * HD + n];
}

// ---------------------------------------------------------------------------
// A-fragment pack (16x32 bf16, contiguous K order, same as B):
//   lane<16 : M=lane,    elems 0..15 = K kb..kb+15 (kb=0)
//   lane>=16: M=lane-16, same with kb=16
// ---------------------------------------------------------------------------
__device__ __forceinline__ v16bf load_a_frag(const bf16_t* row, int k0, int kbase)
{
    const v8bf lo = *(const v8bf*)(row + k0 + kbase);
    const v8bf hi = *(const v8bf*)(row + k0 + kbase + 8);
    v16bf a;
#pragma unroll
    for (int t = 0; t < 8; ++t) { a[t] = lo[t]; a[t + 8] = hi[t]; }
    return a;
}

// ---------------------------------------------------------------------------
// Edge kernel: block = (b, i, 128 j's). Weight-stationary tiling:
//   - e1/e2 are BLOCK-shared (128 x 256 bf16 in LDS)
//   - wave w holds B-fragments of its 2 N-tiles (full K) in 128 VGPRs and
//     sweeps all 8 M-tiles -> each weight matrix is read once per block
//     (128 KB/block vs 1 MB/block before: 8x less L2 traffic, ~128 FLOP/byte)
//   - final 256->1 projection folded into GEMM2 epilogue; cross-wave
//     reduction via shfl + small LDS array (no atomics)
// ---------------------------------------------------------------------------
__global__ __launch_bounds__(256)
void edge_kernel(const float* __restrict__ u,   const float* __restrict__ bc1,
                 const bf16_t* __restrict__ W2t, const float* __restrict__ bc2,
                 const bf16_t* __restrict__ W3t, const float* __restrict__ bc3,
                 const float* __restrict__ Wo,   const float* __restrict__ bo,
                 float* __restrict__ out)
{
    extern __shared__ char smem_raw[];
    bf16_t* e1     = (bf16_t*)smem_raw;                    // [MT][ROWP]
    bf16_t* e2     = e1 + (size_t)MT * ROWP;               // [MT][ROWP]
    float*  osum_w = (float*)(e2 + (size_t)MT * ROWP);     // [8][MT]

    const int tid  = threadIdx.x;
    const int lane = tid & 31;
    const int wave = tid >> 5;
    const int jt   = blockIdx.x;                 // 0..NP/MT-1
    const int i    = blockIdx.y;                 // 0..NP-1
    const int b    = blockIdx.z;                 // 0..BB-1
    const int j0   = jt * MT;

    const float* ub = u + (size_t)b * NP * HD;

    // ---- phase 1: wave w builds e1 rows 16w..16w+15 ----
    {
        const int c0 = lane * 8;
        const float4* uip = (const float4*)(ub + (size_t)i * HD + c0);
        const float4  t0 = uip[0], t1 = uip[1];
        const float4* bp = (const float4*)(bc1 + c0);
        const float4  b0 = bp[0], b1 = bp[1];
        const float ui[8] = { t0.x, t0.y, t0.z, t0.w, t1.x, t1.y, t1.z, t1.w };
        const float bs[8] = { b0.x, b0.y, b0.z, b0.w, b1.x, b1.y, b1.z, b1.w };
#pragma unroll 4
        for (int r = 0; r < 16; ++r) {
            const int m = wave * 16 + r;
            const float4* ujp = (const float4*)(ub + (size_t)(j0 + m) * HD + c0);
            const float4 a0 = ujp[0], a1 = ujp[1];
            const float v[8] = { a0.x, a0.y, a0.z, a0.w, a1.x, a1.y, a1.z, a1.w };
            bf16_t tmp[8];
#pragma unroll
            for (int t = 0; t < 8; ++t) {
                float e = v[t] - ui[t] + bs[t];
                e = e > 0.f ? e : 0.f;
                tmp[t] = (bf16_t)e;
            }
            *(v8bf*)(e1 + (size_t)m * ROWP + c0) = *(const v8bf*)tmp;
        }
    }

    const int nlo   = lane & 15;      // D/B column within tile; A row
    const int khalf = lane >> 4;      // 0/1: K-half selector
    const int kbA   = khalf * 16;     // A-fragment K base
    const int kbB   = khalf * 16;     // B-fragment K base

    // wave's two N-tile columns
    const int nA = wave * 32 + nlo;
    const int nB = nA + 16;

    // ---- preload W2 B-fragments (full K) into registers: weight-stationary ----
    v16bf bw[2][8];
    {
        const bf16_t* wrA = W2t + (size_t)nA * HD + kbB;
        const bf16_t* wrB = W2t + (size_t)nB * HD + kbB;
#pragma unroll
        for (int kt = 0; kt < 8; ++kt) {
            bw[0][kt] = *(const v16bf*)(wrA + kt * 32);
            bw[1][kt] = *(const v16bf*)(wrB + kt * 32);
        }
    }
    const float b2A = bc2[nA], b2B = bc2[nB];

    __syncthreads();   // e1 complete

    // ---- GEMM1: e2 = relu(e1 @ Wc2 + bc2), sweep all 8 M-tiles ----
    for (int mt = 0; mt < 8; ++mt) {
        v16bf afr[8];
        const bf16_t* arow = e1 + (size_t)(mt * 16 + nlo) * ROWP;
#pragma unroll
        for (int kt = 0; kt < 8; ++kt)
            afr[kt] = load_a_frag(arow, kt * 32, kbA);

        v8f acc0 = {}, acc1 = {};
#pragma unroll
        for (int kt = 0; kt < 8; ++kt) {
            acc0 = __builtin_amdgcn_wmma_f32_16x16x32_bf16(
                       false, afr[kt], false, bw[0][kt], (short)0, acc0, false, false);
            acc1 = __builtin_amdgcn_wmma_f32_16x16x32_bf16(
                       false, afr[kt], false, bw[1][kt], (short)0, acc1, false, false);
        }
#pragma unroll
        for (int r = 0; r < 8; ++r) {
            const int m = mt * 16 + khalf * 8 + r;   // D row = VGPR idx + 8*(lane>=16)
            float v0 = acc0[r] + b2A; v0 = v0 > 0.f ? v0 : 0.f;
            float v1 = acc1[r] + b2B; v1 = v1 > 0.f ? v1 : 0.f;
            e2[(size_t)m * ROWP + nA] = (bf16_t)v0;
            e2[(size_t)m * ROWP + nB] = (bf16_t)v1;
        }
    }

    // ---- swap in W3 B-fragments ----
    {
        const bf16_t* wrA = W3t + (size_t)nA * HD + kbB;
        const bf16_t* wrB = W3t + (size_t)nB * HD + kbB;
#pragma unroll
        for (int kt = 0; kt < 8; ++kt) {
            bw[0][kt] = *(const v16bf*)(wrA + kt * 32);
            bw[1][kt] = *(const v16bf*)(wrB + kt * 32);
        }
    }
    const float b3A = bc3[nA], b3B = bc3[nB];
    const float woA = Wo[nA],  woB = Wo[nB];

    __syncthreads();   // e2 complete

    // ---- GEMM2 + output projection ----
    for (int mt = 0; mt < 8; ++mt) {
        v16bf afr[8];
        const bf16_t* arow = e2 + (size_t)(mt * 16 + nlo) * ROWP;
#pragma unroll
        for (int kt = 0; kt < 8; ++kt)
            afr[kt] = load_a_frag(arow, kt * 32, kbA);

        v8f acc0 = {}, acc1 = {};
#pragma unroll
        for (int kt = 0; kt < 8; ++kt) {
            acc0 = __builtin_amdgcn_wmma_f32_16x16x32_bf16(
                       false, afr[kt], false, bw[0][kt], (short)0, acc0, false, false);
            acc1 = __builtin_amdgcn_wmma_f32_16x16x32_bf16(
                       false, afr[kt], false, bw[1][kt], (short)0, acc1, false, false);
        }

        float part[8];
#pragma unroll
        for (int r = 0; r < 8; ++r) {
            float v0 = acc0[r] + b3A; v0 = v0 > 0.f ? v0 : 0.f;
            float v1 = acc1[r] + b3B; v1 = v1 > 0.f ? v1 : 0.f;
            part[r] = v0 * woA + v1 * woB;
        }
        // reduce across the 16 lanes of each half-wave (rows identical per half)
#pragma unroll
        for (int off = 8; off >= 1; off >>= 1) {
#pragma unroll
            for (int r = 0; r < 8; ++r)
                part[r] += __shfl_xor(part[r], off, 16);
        }
        if (nlo == 0) {
            const int m0 = mt * 16 + khalf * 8;
#pragma unroll
            for (int r = 0; r < 8; ++r)
                osum_w[wave * MT + m0 + r] = part[r];
        }
    }

    __syncthreads();   // all per-wave partials in LDS

    // ---- final cross-wave reduction + store ----
    if (tid < MT) {
        float s = bo[0];
#pragma unroll
        for (int w = 0; w < 8; ++w)
            s += osum_w[w * MT + tid];
        out[((size_t)b * NP + i) * NP + j0 + tid] = s;
    }
}

// ---------------------------------------------------------------------------
extern "C" void kernel_launch(void* const* d_in, const int* in_sizes, int n_in,
                              void* d_out, int out_size, void* d_ws, size_t ws_size,
                              hipStream_t stream)
{
    const float* x   = (const float*)d_in[0];
    const float* Wa  = (const float*)d_in[1];
    const float* ba  = (const float*)d_in[2];
    const float* Wb  = (const float*)d_in[3];
    const float* bb  = (const float*)d_in[4];
    const float* Wc1 = (const float*)d_in[5];
    const float* bc1 = (const float*)d_in[6];
    const float* Wc2 = (const float*)d_in[7];
    const float* bc2 = (const float*)d_in[8];
    const float* Wc3 = (const float*)d_in[9];
    const float* bc3 = (const float*)d_in[10];
    const float* Wo  = (const float*)d_in[11];
    const float* bo  = (const float*)d_in[12];
    float* out = (float*)d_out;

    char* ws = (char*)d_ws;
    const size_t MB = 1u << 20;
    float*  h1  = (float*)(ws + 0 * MB);            // [BB*NP, HD] fp32
    float*  h2  = (float*)(ws + 1 * MB);            // [BB*NP, HD] fp32
    float*  u   = (float*)(ws + 2 * MB);            // [BB*NP, HD] fp32
    bf16_t* W2t = (bf16_t*)(ws + 3 * MB);           // [HD, HD] bf16 transposed
    bf16_t* W3t = (bf16_t*)(ws + 3 * MB + (size_t)HD * HD * sizeof(bf16_t));

    const int rows = BB * NP;
    // Node MLP + linearity-folded first edge layer: u = h2 @ Wc1 (NO bias —
    // bc1 is applied once inside the pairwise difference in edge_kernel).
    lin_kernel<<<rows, HD, 0, stream>>>(x,  Wa, ba, h1, 1);
    lin_kernel<<<rows, HD, 0, stream>>>(h1, Wb, bb, h2, 1);
    lin_kernel<<<rows, HD, 0, stream>>>(h2, Wc1, nullptr, u, 0);

    tconv_kernel<<<HD, HD, 0, stream>>>(Wc2, W2t);
    tconv_kernel<<<HD, HD, 0, stream>>>(Wc3, W3t);

    dim3 grid(NP / MT, NP, BB);
    const size_t shmem = (size_t)2 * MT * ROWP * sizeof(bf16_t)   // e1 + e2
                       + (size_t)8 * MT * sizeof(float);          // per-wave partials
    edge_kernel<<<grid, 256, shmem, stream>>>(u, bc1, W2t, bc2, W3t, bc3, Wo, bo, out);
}
